// hierEncoder_27633819583223
// MI455X (gfx1250) — matrix operand order repaired and checked by reading, not compiled, not run on hardware
//
#include <hip/hip_runtime.h>
#include <hip/hip_bf16.h>
#include <math.h>

#define HDIM 512
#define TLEN 4096
#define G3   1536   /* 3*HDIM */

#define NSTRIP 48                 /* B panel columns per block (3 wmma tiles) */
#define BROWSTRIDE 520            /* 512 + 8 bf16 pad -> kills LDS bank conflicts */

typedef __attribute__((ext_vector_type(16))) __bf16 v16bf;
typedef __attribute__((ext_vector_type(8)))  __bf16 v8bf;
typedef __attribute__((ext_vector_type(8)))  float  v8f;
typedef __attribute__((ext_vector_type(4)))  unsigned int v4u;
typedef __attribute__((ext_vector_type(8)))  int v8i;
typedef __attribute__((ext_vector_type(4)))  int v4i;

__device__ __forceinline__ __bf16 f2bf(float f) {
  union { float f; unsigned u; } a; a.f = f;
  unsigned r = a.u + 0x7FFFu + ((a.u >> 16) & 1u);   // round-to-nearest-even
  union { unsigned short s; __bf16 b; } o; o.s = (unsigned short)(r >> 16);
  return o.b;
}

__device__ __forceinline__ float sigmoidf_(float x) {
  return 1.0f / (1.0f + __expf(-x));
}

// ---------------------------------------------------------------------------
// Kernel 1: one-shot conversions (done ONCE, so the WMMA GEMM never touches
// fp32->bf16 VALU work):
//   Wb[n][k]      = bf16(W_ih1[n][k])                        (1.5 MB)
//   Xbf[s][t][k]  = bf16(emb[tok_s[t]][k])   (gather+convert, 8 MB, L2-hot)
//   hbuf          = 0
// ---------------------------------------------------------------------------
__global__ void prep_kernel(const int* __restrict__ tokx,
                            const int* __restrict__ toky,
                            const float* __restrict__ emb,
                            const float* __restrict__ Wih1,
                            __bf16* __restrict__ Wb,
                            __bf16* __restrict__ Xbf,
                            float* __restrict__ hbuf) {
  int tid = blockIdx.x * blockDim.x + threadIdx.x;
  int stride = gridDim.x * blockDim.x;
  for (int i = tid; i < G3 * HDIM; i += stride) Wb[i] = f2bf(Wih1[i]);
  for (int i = tid; i < 2 * TLEN * HDIM; i += stride) {
    int k = i & (HDIM - 1);
    int tg = i >> 9;                // global row 0..8191
    int s = tg >> 12;
    int t = tg & (TLEN - 1);
    int tok = s ? toky[t] : tokx[t];
    Xbf[i] = f2bf(emb[(long)tok * HDIM + k]);
  }
  if (tid < 2 * 2 * HDIM) hbuf[tid] = 0.0f;      // [seq][phase][HDIM]
}

// ---------------------------------------------------------------------------
// Kernel 2: G[s][t][n] = sum_k Xbf[s][t][k]*Wb[n][k] + b_ih1[n]  via WMMA.
// Block = 256 threads (8 waves). Block computes a 128 x 48 output panel:
//   wave w -> rows (mg*8+w)*16 .. +15, cols ng*48 .. +47 (3 accumulators).
// B panel (48 rows x 512 k, bf16) staged into LDS by the Tensor Data Mover
// with D#.pad (16B after each 1024B) -> row stride 520 elems, conflict-free
// 16-lane tile reads. A operands are direct b128 loads from bf16 X.
// ---------------------------------------------------------------------------
__global__ __launch_bounds__(256) void gi_gemm_kernel(
    const __bf16* __restrict__ Xbf, const __bf16* __restrict__ Wb,
    const float* __restrict__ bih, float* __restrict__ G) {
  __shared__ __bf16 Bs[NSTRIP * BROWSTRIDE];    // 49,920 B

  const int NGRP = G3 / NSTRIP;                 // 32 col groups
  const int MGRP = TLEN / (8 * 16);             // 32 row groups per sequence
  int b = blockIdx.x;                           // 2 * 32 * 32 = 2048 blocks
  int s = b / (MGRP * NGRP);
  int r = b - s * (MGRP * NGRP);
  int mg = r / NGRP, ng = r % NGRP;
  int n0 = ng * NSTRIP;

  int tid  = (int)threadIdx.x;
  int wid  = tid >> 5;
  int lane = tid & 31;

  // ---- stage B panel: W_ih1 bf16 rows [n0, n0+48) x 512 k into LDS --------
#if defined(__has_builtin) && __has_builtin(__builtin_amdgcn_tensor_load_to_lds)
  if (wid == 0) {
    unsigned long long ga =
        (unsigned long long)(uintptr_t)(Wb + (long)n0 * HDIM);
    unsigned int lds = (unsigned int)(uintptr_t)&Bs[0];
    v4u g0 = { 1u,                                   // count=1 (valid D#)
               lds,                                  // lds_addr
               (unsigned int)ga,                     // global_addr[31:0]
               ((unsigned int)(ga >> 32) & 0x01FFFFFFu) | (2u << 30) }; // +type=2
    union { unsigned int d[8]; v8i v; } g1u;
    g1u.d[0] = (1u << 16)      // data_size = 2B
             | (1u << 20)      // pad_enable
             | (7u << 22)      // pad_interval: every 256 dwords (1024B)
             | (3u << 25);     // pad_amount : 4 dwords (16B)
    g1u.d[1] = (unsigned)(HDIM & 0xFFFF) << 16;      // tensor_dim0 lo16
    g1u.d[2] = ((unsigned)HDIM >> 16) | ((unsigned)G3 << 16); // dim0 hi | dim1 lo
    g1u.d[3] = ((unsigned)G3 >> 16) | ((unsigned)HDIM << 16); // dim1 hi | tile_dim0
    g1u.d[4] = (unsigned)NSTRIP;                     // tile_dim1 | tile_dim2=0
    g1u.d[5] = (unsigned)HDIM;                       // tensor_dim0_stride lo32
    g1u.d[6] = 0u;
    g1u.d[7] = 0u;
    v4i z4 = {0, 0, 0, 0};
#if __has_include(<hip/amd_detail/amd_gfx1250_TDM.h>)
    v8i z8 = {0, 0, 0, 0, 0, 0, 0, 0};
    __builtin_amdgcn_tensor_load_to_lds(g0, g1u.v, z4, z4, z8, 0);
#else
    __builtin_amdgcn_tensor_load_to_lds(g0, g1u.v, z4, z4, 0);
#endif
    __builtin_amdgcn_s_wait_tensorcnt(0);
  }
#else
  for (int i = tid; i < NSTRIP * HDIM; i += 256) {
    int row = i >> 9, col = i & (HDIM - 1);
    Bs[row * BROWSTRIDE + col] = Wb[(long)(n0 + row) * HDIM + col];
  }
#endif
  __syncthreads();

  int half = lane >> 4;
  int l15  = lane & 15;
  int mt   = mg * 8 + wid;
  int m    = mt * 16 + l15;
  const __bf16* arow = Xbf + ((long)s * TLEN + m) * HDIM;

  v8f c0 = {0.f,0.f,0.f,0.f,0.f,0.f,0.f,0.f};
  v8f c1 = c0, c2 = c0;
  for (int kb = 0; kb < HDIM; kb += 32) {
    // A (16x32): lane = row m, K = {ka..ka+15}, ka=kb+half*16
    union { v16bf v; v8bf h[2]; } A;
    int ka = kb + half * 16;
    A.h[0] = *(const v8bf*)(arow + ka);
    A.h[1] = *(const v8bf*)(arow + ka + 8);
    // B (32x16) tiles from LDS: lane = col, 16 contiguous K at kb+half*16
#pragma unroll
    for (int tt = 0; tt < 3; ++tt) {
      const __bf16* bp = &Bs[(tt * 16 + l15) * BROWSTRIDE + kb + half * 16];
      v16bf Bv = *(const v16bf*)bp;
      v8f acc = tt == 0 ? c0 : (tt == 1 ? c1 : c2);
      acc = __builtin_amdgcn_wmma_f32_16x16x32_bf16(
          false, A.v, false, Bv, (short)0, acc, false, false);
      if (tt == 0) c0 = acc; else if (tt == 1) c1 = acc; else c2 = acc;
    }
  }
  // store: lane covers col tt*16+l15; VGPR v -> row mt*16 + v + 8*half
  float* Gs = G + (long)s * TLEN * G3;
#pragma unroll
  for (int tt = 0; tt < 3; ++tt) {
    int n = n0 + tt * 16 + l15;
    float bias = bih[n];
    v8f acc = tt == 0 ? c0 : (tt == 1 ? c1 : c2);
#pragma unroll
    for (int v = 0; v < 8; ++v) {
      int mm = mt * 16 + v + 8 * half;
      Gs[(long)mm * G3 + n] = acc[v] + bias;
    }
  }
}

// ---------------------------------------------------------------------------
// Kernel 3: sequential GRU recurrence, cluster-parallel matvec.
// Grid = 32 WGs (intended: 2 clusters of 16, one per sequence).
// WG (s, part) owns gates j in [32*part, 32*part+32); thread t:
//   j = 32*part + (t>>4), k-chunk = (t&15)*32. The 96 W_hh weights per
//   thread live in VGPRs for all 4096 steps (3MB/seq register-resident
//   across the cluster). h is exchanged via agent-scope L2 + cluster barrier.
// ---------------------------------------------------------------------------
__global__ __launch_bounds__(512) void gru1_recurrence_kernel(
    const float* __restrict__ Whh, const float* __restrict__ bhh,
    const float* __restrict__ G, float* hbuf) {
  int wg = blockIdx.x;
  int s = wg >> 4;                 // sequence 0/1 (one cluster each)
  int part = wg & 15;
  int t = (int)threadIdx.x;
  int jl = t >> 4;                 // 0..31 local gate
  int j  = part * 32 + jl;         // 0..511 global gate index
  int kc = (t & 15) * 32;          // k-chunk base

  float wr[32], wz[32], wn[32];
#pragma unroll
  for (int i = 0; i < 32; ++i) wr[i] = Whh[(long)j * HDIM + kc + i];
#pragma unroll
  for (int i = 0; i < 32; ++i) wz[i] = Whh[(long)(j + HDIM) * HDIM + kc + i];
#pragma unroll
  for (int i = 0; i < 32; ++i) wn[i] = Whh[(long)(j + 2 * HDIM) * HDIM + kc + i];
  float br = bhh[j], bz = bhh[j + HDIM], bn = bhh[j + 2 * HDIM];

  const float* Gs = G + (long)s * TLEN * G3;
  for (int step = 0; step < TLEN; ++step) {
    int rp = step & 1, wp = rp ^ 1;
    float* hR = hbuf + (s * 2 + rp) * HDIM;
    float* hW = hbuf + (s * 2 + wp) * HDIM;

    float h[32];
#pragma unroll
    for (int i = 0; i < 32; ++i)
      h[i] = __hip_atomic_load(&hR[kc + i], __ATOMIC_RELAXED,
                               __HIP_MEMORY_SCOPE_AGENT);
    float ar = 0.f, az = 0.f, an = 0.f;
#pragma unroll
    for (int i = 0; i < 32; ++i) {
      ar = fmaf(h[i], wr[i], ar);
      az = fmaf(h[i], wz[i], az);
      an = fmaf(h[i], wn[i], an);
    }
    // reduce the 16 k-chunks (lanes sharing t>>4 live in one wave half)
#pragma unroll
    for (int o = 1; o < 16; o <<= 1) {
      ar += __shfl_xor(ar, o, 32);
      az += __shfl_xor(az, o, 32);
      an += __shfl_xor(an, o, 32);
    }
    if ((t & 15) == 0) {
      const float* g = Gs + (long)step * G3;
      if (step + 1 < TLEN) __builtin_prefetch(g + G3, 0, 0); // next step from L2
      float r = sigmoidf_(g[j] + ar + br);
      float z = sigmoidf_(g[HDIM + j] + az + bz);
      float nn = tanhf(g[2 * HDIM + j] + r * (an + bn));
      float hj = __hip_atomic_load(&hR[j], __ATOMIC_RELAXED,
                                   __HIP_MEMORY_SCOPE_AGENT);
      float hnew = (1.0f - z) * nn + z * hj;
      __hip_atomic_store(&hW[j], hnew, __ATOMIC_RELAXED,
                         __HIP_MEMORY_SCOPE_AGENT);
    }
    __builtin_amdgcn_fence(__ATOMIC_RELEASE, "agent");
    __syncthreads();
    __builtin_amdgcn_s_cluster_barrier();   // s_barrier_signal/wait -3
    __builtin_amdgcn_fence(__ATOMIC_ACQUIRE, "agent");
  }
}

// ---------------------------------------------------------------------------
// Kernel 4: gru2 (2 steps over [hx, hy]) + FC 512->128 relu -> 128->2 relu
//           -> log_softmax. One 512-thread block; thread j owns gate j.
// Final gru1 hiddens live in hbuf[s][phase 0] (TLEN is even).
// ---------------------------------------------------------------------------
__global__ __launch_bounds__(512) void gru2_head_kernel(
    const float* __restrict__ Wih2, const float* __restrict__ Whh2,
    const float* __restrict__ bih2, const float* __restrict__ bhh2,
    const float* __restrict__ W1, const float* __restrict__ b1,
    const float* __restrict__ W2, const float* __restrict__ b2,
    const float* __restrict__ hbuf, float* __restrict__ out) {
  __shared__ float h[HDIM];
  __shared__ float xin[HDIM];
  __shared__ float o1[128];
  int t = (int)threadIdx.x;
  h[t] = 0.0f;
  __syncthreads();
  for (int step = 0; step < 2; ++step) {
    xin[t] = __hip_atomic_load((float*)&hbuf[(step * 2 + 0) * HDIM + t],
                               __ATOMIC_RELAXED, __HIP_MEMORY_SCOPE_AGENT);
    __syncthreads();
    float gi[3], gh[3];
#pragma unroll
    for (int gidx = 0; gidx < 3; ++gidx) {
      const float* wi = Wih2 + (long)(gidx * HDIM + t) * HDIM;
      const float* wh = Whh2 + (long)(gidx * HDIM + t) * HDIM;
      float si = 0.f, sh = 0.f;
      for (int k = 0; k < HDIM; ++k) {
        si = fmaf(xin[k], wi[k], si);
        sh = fmaf(h[k], wh[k], sh);
      }
      gi[gidx] = si + bih2[gidx * HDIM + t];
      gh[gidx] = sh + bhh2[gidx * HDIM + t];
    }
    float r = sigmoidf_(gi[0] + gh[0]);
    float z = sigmoidf_(gi[1] + gh[1]);
    float nn = tanhf(gi[2] + r * gh[2]);
    float hnew = (1.0f - z) * nn + z * h[t];
    __syncthreads();
    h[t] = hnew;
    __syncthreads();
  }
  if (t < 128) {
    const float* w = W1 + (long)t * HDIM;
    float acc = b1[t];
    for (int k = 0; k < HDIM; ++k) acc = fmaf(h[k], w[k], acc);
    o1[t] = fmaxf(acc, 0.0f);
  }
  __syncthreads();
  if (t == 0) {
    float z0 = b2[0], z1 = b2[1];
    for (int k = 0; k < 128; ++k) {
      z0 = fmaf(o1[k], W2[k], z0);
      z1 = fmaf(o1[k], W2[128 + k], z1);
    }
    z0 = fmaxf(z0, 0.0f); z1 = fmaxf(z1, 0.0f);
    float mx = fmaxf(z0, z1);
    float lse = mx + __logf(__expf(z0 - mx) + __expf(z1 - mx));
    out[0] = z0 - lse;
    out[1] = z1 - lse;
  }
}

// ---------------------------------------------------------------------------
extern "C" void kernel_launch(void* const* d_in, const int* in_sizes, int n_in,
                              void* d_out, int out_size, void* d_ws, size_t ws_size,
                              hipStream_t stream) {
  (void)in_sizes; (void)n_in; (void)out_size; (void)ws_size;
  const int*   x    = (const int*)d_in[0];
  const int*   y    = (const int*)d_in[1];
  const float* emb  = (const float*)d_in[2];
  const float* Wih1 = (const float*)d_in[3];
  const float* Whh1 = (const float*)d_in[4];
  const float* bih1 = (const float*)d_in[5];
  const float* bhh1 = (const float*)d_in[6];
  const float* Wih2 = (const float*)d_in[7];
  const float* Whh2 = (const float*)d_in[8];
  const float* bih2 = (const float*)d_in[9];
  const float* bhh2 = (const float*)d_in[10];
  const float* W1   = (const float*)d_in[11];
  const float* b1   = (const float*)d_in[12];
  const float* W2   = (const float*)d_in[13];
  const float* b2   = (const float*)d_in[14];

  char* ws = (char*)d_ws;
  const size_t WB_BYTES = (size_t)G3 * HDIM * 2;                 // 1.5 MB
  const size_t X_OFF    = (WB_BYTES + 255) & ~(size_t)255;
  const size_t X_BYTES  = (size_t)2 * TLEN * HDIM * 2;           // 8 MB bf16 X
  const size_t G_OFF    = (X_OFF + X_BYTES + 255) & ~(size_t)255;
  const size_t G_BYTES  = (size_t)2 * TLEN * G3 * 4;             // 50 MB gi
  const size_t H_OFF    = G_OFF + G_BYTES;

  __bf16* Wb  = (__bf16*)ws;
  __bf16* Xbf = (__bf16*)(ws + X_OFF);
  float*  G   = (float*)(ws + G_OFF);
  float*  hbf = (float*)(ws + H_OFF);                            // [2][2][HDIM]

  prep_kernel<<<512, 256, 0, stream>>>(x, y, emb, Wih1, Wb, Xbf, hbf);

  // 2 seqs * 32 row-groups * 32 col-groups = 2048 blocks of 8 waves
  gi_gemm_kernel<<<2048, 256, 0, stream>>>(Xbf, Wb, bih1, G);

  // intended dispatch: 2 clusters x 16 workgroups (one cluster per sequence)
  gru1_recurrence_kernel<<<32, 512, 0, stream>>>(Whh1, bhh1, G, hbf);

  gru2_head_kernel<<<1, 512, 0, stream>>>(Wih2, Whh2, bih2, bhh2,
                                          W1, b1, W2, b2, hbf, (float*)d_out);
}
